// RWKV6TimeMixing_8452495638552
// MI455X (gfx1250) — matrix operand order repaired and checked by reading, not compiled, not run on hardware
//
#include <hip/hip_runtime.h>
#include <hip/hip_bf16.h>

typedef __attribute__((ext_vector_type(16))) _Float16     v16h;
typedef __attribute__((ext_vector_type(8)))  float        v8f;
typedef __attribute__((ext_vector_type(2)))  float        v2f;
typedef __attribute__((ext_vector_type(4)))  unsigned int v4u;
typedef __attribute__((ext_vector_type(8)))  int          v8i;
typedef __attribute__((ext_vector_type(4)))  int          v4i;

#define D_MODEL 1024
#define T_SEQ   2048
#define NROWS   4096   // B*T
#define H_HEADS 16
#define DHEAD   64
#define CLEN    16     // chunk length
#define NCHUNK  (T_SEQ / CLEN)          // 128
#define NTASK   (2 * H_HEADS * NCHUNK)  // B*H*NCHUNK = 4096

#if __has_builtin(__builtin_amdgcn_tensor_load_to_lds) && __has_builtin(__builtin_amdgcn_s_wait_tensorcnt)
#define HAVE_TDM 1
#else
#define HAVE_TDM 0
#endif

__device__ __forceinline__ float sigmoidf_(float z) { return 1.0f / (1.0f + __expf(-z)); }

// epi: 0=id, 1=sigmoid(tanh(z)), 2=silu(z), 3=-softplus(-z)-0.5
__device__ __forceinline__ float apply_epi(float z, int epi) {
    if (epi == 1) return sigmoidf_(tanhf(z));
    if (epi == 2) return z * sigmoidf_(z);
    if (epi == 3) return -__logf(1.0f + __expf(-z)) - 0.5f;
    return z;
}

// f32 16x16x4 WMMA (A:16x4 -> 2 VGPR, B:4x16 -> 2 VGPR)
__device__ __forceinline__ v8f wmma4(float a0, float a1, float b0, float b1, v8f c) {
#if __has_builtin(__builtin_amdgcn_wmma_f32_16x16x4_f32)
    v2f a = {a0, a1};
    v2f b = {b0, b1};
    return __builtin_amdgcn_wmma_f32_16x16x4_f32(false, a, false, b, (short)0, c, false, false);
#else
    #pragma unroll
    for (int i = 0; i < 8; ++i) c[i] = fmaf(a0, b0, fmaf(a1, b1, c[i]));  // placeholder
    return c;
#endif
}

// ---------------------------------------------------------------------------
// one-time weight conversion f32 -> f16 (row-major [N,K] preserved)
// ---------------------------------------------------------------------------
__global__ __launch_bounds__(256)
void conv_f32_to_f16(const float* __restrict__ src, _Float16* __restrict__ dst)
{
    const int i = blockIdx.x * blockDim.x + threadIdx.x;   // one float4 each
    float4 f = reinterpret_cast<const float4*>(src)[i];
    alignas(8) _Float16 h[4] = {(_Float16)f.x, (_Float16)f.y, (_Float16)f.z, (_Float16)f.w};
    reinterpret_cast<float2*>(dst)[i] = *reinterpret_cast<const float2*>(h);
}

// ---------------------------------------------------------------------------
// C = A_eff x Wh^T, Wh is f16 [N,K] row-major.
// asrc: 0 = token-shifted x; 1 = lerp(x, x_shift; mix, dd); 2 = A0 directly.
// B tile (64 rows x 32 halves) is DMA'd by the Tensor Data Mover with LDS
// padding (16 DWORDs data + 4 DWORDs pad == 40-half row stride).
// ---------------------------------------------------------------------------
__global__ __launch_bounds__(128)
void rwkv_gemm_wmma(const float* __restrict__ A0,
                    const float* __restrict__ dd,
                    const float* __restrict__ mix,
                    const _Float16* __restrict__ Wh,
                    float* __restrict__ out,
                    int asrc, int epi)
{
    __shared__ _Float16 Alds[64 * 40];
    __shared__ _Float16 Blds[64 * 40];

    const int tid  = threadIdx.x;
    const int wave = tid >> 5;
    const int lane = tid & 31;
    const int m0 = blockIdx.y * 64;
    const int n0 = blockIdx.x * 64;

    v8f acc[4] = {};

    const int srow = tid >> 1;
    const int kseg = (tid & 1) << 4;

    for (int kc = 0; kc < D_MODEL; kc += 32) {
#if HAVE_TDM
        if (wave == 0) {
            const unsigned ldsb = (unsigned)(uintptr_t)(&Blds[0]);
            const unsigned long long ga =
                (unsigned long long)(uintptr_t)(Wh + (size_t)n0 * D_MODEL + kc);
            // D# group0: count=1, lds_addr, global_addr[56:0], type=2
            v4u g0 = { 1u, ldsb, (unsigned)ga,
                       (unsigned)(((ga >> 32) & 0x1FFFFFFull)) | (2u << 30) };
            // D# group1: data_size=2B, pad_enable, pad_interval=16 DW, pad_amount=4 DW,
            //            tensor_dim0=1024, tensor_dim1=1024, tile_dim0=32, tile_dim1=64,
            //            tensor_dim0_stride=1024
            v8i g1 = { (int)((1u << 16) | (1u << 20) | (3u << 22) | (3u << 25)),
                       (int)(1024u << 16),            // tensor_dim0[15:0] in [63:48]
                       (int)(1024u << 16),            // tensor_dim0 hi=0 | tensor_dim1[15:0]
                       (int)(32u << 16),              // tensor_dim1 hi=0 | tile_dim0=32
                       64,                            // tile_dim1=64, tile_dim2=0
                       1024,                          // tensor_dim0_stride lo
                       0, 0 };
            v4i z4 = {0, 0, 0, 0};
            v8i z8 = {0, 0, 0, 0, 0, 0, 0, 0};
            __builtin_amdgcn_tensor_load_to_lds(g0, g1, z4, z4, z8, 0);
        }
#endif
        {   // stage A tile (fused token-shift / ddlerp mix), f32 -> f16
            const int mg = m0 + srow;
            const int t  = mg & (T_SEQ - 1);
            float av[16];
            if (asrc == 2) {
                const float4* p = reinterpret_cast<const float4*>(A0 + (size_t)mg * D_MODEL + kc + kseg);
                #pragma unroll
                for (int q = 0; q < 4; ++q) {
                    float4 f = p[q];
                    av[4*q+0] = f.x; av[4*q+1] = f.y; av[4*q+2] = f.z; av[4*q+3] = f.w;
                }
            } else {
                float xs[16];
                if (t == 0) {
                    #pragma unroll
                    for (int i = 0; i < 16; ++i) xs[i] = 0.0f;
                } else {
                    const float4* ps = reinterpret_cast<const float4*>(A0 + (size_t)(mg - 1) * D_MODEL + kc + kseg);
                    #pragma unroll
                    for (int q = 0; q < 4; ++q) {
                        float4 f = ps[q];
                        xs[4*q+0] = f.x; xs[4*q+1] = f.y; xs[4*q+2] = f.z; xs[4*q+3] = f.w;
                    }
                }
                if (asrc == 0) {
                    #pragma unroll
                    for (int i = 0; i < 16; ++i) av[i] = xs[i];
                } else {
                    const float4* px = reinterpret_cast<const float4*>(A0 + (size_t)mg * D_MODEL + kc + kseg);
                    const float4* pd = reinterpret_cast<const float4*>(dd + (size_t)mg * D_MODEL + kc + kseg);
                    const float4* pm = reinterpret_cast<const float4*>(mix + kc + kseg);
                    float xv[16], dv[16], mv[16];
                    #pragma unroll
                    for (int q = 0; q < 4; ++q) {
                        float4 fx = px[q], fd = pd[q], fm = pm[q];
                        xv[4*q+0] = fx.x; xv[4*q+1] = fx.y; xv[4*q+2] = fx.z; xv[4*q+3] = fx.w;
                        dv[4*q+0] = fd.x; dv[4*q+1] = fd.y; dv[4*q+2] = fd.z; dv[4*q+3] = fd.w;
                        mv[4*q+0] = fm.x; mv[4*q+1] = fm.y; mv[4*q+2] = fm.z; mv[4*q+3] = fm.w;
                    }
                    #pragma unroll
                    for (int i = 0; i < 16; ++i) {
                        const float m = mv[i] + dv[i] * (1.0f - mv[i]);
                        av[i] = xv[i] * m + xs[i] * (1.0f - m);
                    }
                }
            }
            alignas(16) _Float16 hbuf[16];
            #pragma unroll
            for (int i = 0; i < 16; ++i) hbuf[i] = (_Float16)av[i];
            float4* dst = reinterpret_cast<float4*>(&Alds[srow * 40 + kseg]);
            dst[0] = reinterpret_cast<const float4*>(hbuf)[0];
            dst[1] = reinterpret_cast<const float4*>(hbuf)[1];
        }
#if !HAVE_TDM
        {   // fallback: manual B tile copy (already f16)
            const int col = n0 + srow;
            const _Float16* srcb = Wh + (size_t)col * D_MODEL + kc + kseg;
            const float4* ps = reinterpret_cast<const float4*>(srcb);
            float4* dst = reinterpret_cast<float4*>(&Blds[srow * 40 + kseg]);
            dst[0] = ps[0];
            dst[1] = ps[1];
            if (kc + 32 < D_MODEL)
                __builtin_prefetch(srcb + 32, 0, 3);
        }
#else
        if (wave == 0) __builtin_amdgcn_s_wait_tensorcnt((short)0);
#endif
        __syncthreads();
        {
            const int arow = (wave << 4) + (lane & 15);
            const _Float16* ap = &Alds[arow * 40 + ((lane < 16) ? 0 : 8)];
            v16h a;
            #pragma unroll
            for (int i = 0; i < 8; ++i) { a[i] = ap[i]; a[8 + i] = ap[16 + i]; }
            #pragma unroll
            for (int j = 0; j < 4; ++j) {
                const int bcol = (j << 4) + (lane & 15);
                const _Float16* bp = &Blds[bcol * 40 + ((lane < 16) ? 0 : 8)];
                v16h b;
                #pragma unroll
                for (int i = 0; i < 8; ++i) { b[i] = bp[i]; b[8 + i] = bp[16 + i]; }
                acc[j] = __builtin_amdgcn_wmma_f32_16x16x32_f16(
                    false, a, false, b, (short)0, acc[j], false, false);
            }
        }
        __syncthreads();
    }

    const int rbase = m0 + (wave << 4) + ((lane < 16) ? 0 : 8);
    const int cbase = n0 + (lane & 15);
    #pragma unroll
    for (int j = 0; j < 4; ++j) {
        #pragma unroll
        for (int i = 0; i < 8; ++i) {
            out[(size_t)(rbase + i) * D_MODEL + cbase + (j << 4)] = apply_epi(acc[j][i], epi);
        }
    }
}

// ---------------------------------------------------------------------------
// WKV phase A: D_t = exp(cumsum w), K' = k*exp(-cumsum w), D_end per chunk.
// ---------------------------------------------------------------------------
__global__ __launch_bounds__(64)
void rwkv_wkv_prep(const float* __restrict__ w, const float* __restrict__ kb,
                   float* __restrict__ kp, float* __restrict__ dt, float* __restrict__ dend)
{
    const int id = blockIdx.x;          // bh*NCHUNK + c
    const int bh = id >> 7;
    const int cc = id & (NCHUNK - 1);
    const int b  = bh >> 4;
    const int h  = bh & 15;
    const int k  = threadIdx.x;

    const size_t rowg = (size_t)b * T_SEQ + cc * CLEN;
    float cacc = 0.0f;
    for (int t = 0; t < CLEN; ++t) {
        const size_t gi = (rowg + t) * D_MODEL + h * DHEAD + k;
        cacc += w[gi];
        const size_t ci = ((size_t)id * CLEN + t) * DHEAD + k;
        dt[ci] = __expf(cacc);
        kp[ci] = kb[gi] * __expf(-cacc);
    }
    dend[(size_t)id * DHEAD + k] = __expf(cacc);
}

// ---------------------------------------------------------------------------
// WKV phase B: sequential over chunks per (b,h); 64x64 state in padded LDS.
// ---------------------------------------------------------------------------
__global__ __launch_bounds__(256)
void rwkv_wkv_state(const float* __restrict__ rb, const float* __restrict__ vb,
                    const float* __restrict__ kp, const float* __restrict__ dt,
                    const float* __restrict__ dend, float* __restrict__ ob)
{
    __shared__ float state[64 * 65];
    __shared__ float Rl[16 * 68];
    __shared__ float Vl[16 * 68];
    __shared__ float Kl[16 * 68];
    __shared__ float Dtl[16 * 68];
    __shared__ float Dendl[64];

    const int bh = blockIdx.x;
    const int b  = bh >> 4;
    const int h  = bh & 15;
    const int tid  = threadIdx.x;
    const int wave = tid >> 5;
    const int lane = tid & 31;
    const int lh   = lane & 15;
    const int sel  = (lane < 16) ? 0 : 2;
    const int rofs = (lane < 16) ? 0 : 8;

    for (int idx = tid; idx < 64 * 65; idx += 256) state[idx] = 0.0f;
    __syncthreads();

    for (int c = 0; c < NCHUNK; ++c) {
        const int id = bh * NCHUNK + c;
        const size_t rowg0 = (size_t)b * T_SEQ + c * CLEN;
        {
            const int t  = tid >> 4;
            const int c4 = (tid & 15) << 2;
            const size_t g  = (rowg0 + t) * D_MODEL + h * DHEAD + c4;
            const size_t cb = ((size_t)id * CLEN + t) * DHEAD + c4;
            *reinterpret_cast<float4*>(&Rl[t * 68 + c4])  = *reinterpret_cast<const float4*>(&rb[g]);
            *reinterpret_cast<float4*>(&Vl[t * 68 + c4])  = *reinterpret_cast<const float4*>(&vb[g]);
            *reinterpret_cast<float4*>(&Kl[t * 68 + c4])  = *reinterpret_cast<const float4*>(&kp[cb]);
            *reinterpret_cast<float4*>(&Dtl[t * 68 + c4]) = *reinterpret_cast<const float4*>(&dt[cb]);
            if (tid < 64) Dendl[tid] = dend[(size_t)id * DHEAD + tid];
        }
        __syncthreads();

        if (wave < 4) {  // o_cross: M=16 (t), N=16 k-cols, K=64 (v)
            v8f acc = {};
            #pragma unroll
            for (int kk = 0; kk < 16; ++kk) {
                const int v0 = kk * 4 + sel;
                const float a0 = Rl[lh * 68 + v0];
                const float a1 = Rl[lh * 68 + v0 + 1];
                const float b0 = state[(wave * 16 + lh) * 65 + v0];
                const float b1 = state[(wave * 16 + lh) * 65 + v0 + 1];
                acc = wmma4(a0, a1, b0, b1, acc);
            }
            #pragma unroll
            for (int i = 0; i < 8; ++i) {
                const int row  = i + rofs;
                const int colk = wave * 16 + lh;
                ob[(rowg0 + row) * D_MODEL + h * DHEAD + colk] = acc[i] * Dtl[row * 68 + colk];
            }
        }
        __syncthreads();

        #pragma unroll
        for (int p = 0; p < 2; ++p) {   // state = D_end * (state + K'^T V)
            const int tile = wave * 2 + p;
            const int kt = tile >> 2, vt = tile & 3;
            v8f acc;
            #pragma unroll
            for (int i = 0; i < 8; ++i)
                acc[i] = state[(kt * 16 + i + rofs) * 65 + vt * 16 + lh];
            #pragma unroll
            for (int ss = 0; ss < 4; ++ss) {
                const int s0 = ss * 4 + sel;
                const float a0 = Kl[s0 * 68 + kt * 16 + lh];
                const float a1 = Kl[(s0 + 1) * 68 + kt * 16 + lh];
                const float b0 = Vl[s0 * 68 + vt * 16 + lh];
                const float b1 = Vl[(s0 + 1) * 68 + vt * 16 + lh];
                acc = wmma4(a0, a1, b0, b1, acc);
            }
            #pragma unroll
            for (int i = 0; i < 8; ++i) {
                const int krow = kt * 16 + i + rofs;
                state[krow * 65 + vt * 16 + lh] = acc[i] * Dendl[krow];
            }
        }
        __syncthreads();
    }
}

// ---------------------------------------------------------------------------
// WKV phase C: intra-chunk term, one wave32 per chunk (barrier-light).
// ---------------------------------------------------------------------------
__global__ __launch_bounds__(32)
void rwkv_wkv_intra(const float* __restrict__ rb, const float* __restrict__ vb,
                    const float* __restrict__ kp, const float* __restrict__ dt,
                    float* __restrict__ ob)
{
    __shared__ float Rl[16 * 68];
    __shared__ float Vl[16 * 68];
    __shared__ float Kl[16 * 68];
    __shared__ float Dtl[16 * 68];
    __shared__ float Pl[16 * 20];

    const int id = blockIdx.x;
    const int bh = id >> 7;
    const int cc = id & (NCHUNK - 1);
    const int b  = bh >> 4;
    const int h  = bh & 15;
    const int lane = threadIdx.x;
    const int lh   = lane & 15;
    const int sel  = (lane < 16) ? 0 : 2;
    const int rofs = (lane < 16) ? 0 : 8;
    const size_t rowg0 = (size_t)b * T_SEQ + cc * CLEN;

    for (int idx = lane; idx < 256; idx += 32) {
        const int t  = idx >> 4;
        const int c4 = (idx & 15) << 2;
        const size_t g  = (rowg0 + t) * D_MODEL + h * DHEAD + c4;
        const size_t cb = ((size_t)id * CLEN + t) * DHEAD + c4;
        *reinterpret_cast<float4*>(&Rl[t * 68 + c4])  = *reinterpret_cast<const float4*>(&rb[g]);
        *reinterpret_cast<float4*>(&Vl[t * 68 + c4])  = *reinterpret_cast<const float4*>(&vb[g]);
        *reinterpret_cast<float4*>(&Kl[t * 68 + c4])  = *reinterpret_cast<const float4*>(&kp[cb]);
        *reinterpret_cast<float4*>(&Dtl[t * 68 + c4]) = *reinterpret_cast<const float4*>(&dt[cb]);
    }
    __syncthreads();

    v8f accp = {};
    #pragma unroll
    for (int kk = 0; kk < 16; ++kk) {   // P = R @ V^T
        const int v0 = kk * 4 + sel;
        const float a0 = Rl[lh * 68 + v0];
        const float a1 = Rl[lh * 68 + v0 + 1];
        const float b0 = Vl[lh * 68 + v0];
        const float b1 = Vl[lh * 68 + v0 + 1];
        accp = wmma4(a0, a1, b0, b1, accp);
    }
    #pragma unroll
    for (int i = 0; i < 8; ++i) {       // causal mask, spill P to LDS
        const int row = i + rofs;
        Pl[row * 20 + lh] = (lh <= row) ? accp[i] : 0.0f;
    }
    __syncthreads();

    #pragma unroll
    for (int jt = 0; jt < 4; ++jt) {    // o += D_t * (P @ K')
        v8f acc = {};
        #pragma unroll
        for (int ss = 0; ss < 4; ++ss) {
            const int s0 = ss * 4 + sel;
            const float a0 = Pl[lh * 20 + s0];
            const float a1 = Pl[lh * 20 + s0 + 1];
            const float b0 = Kl[s0 * 68 + jt * 16 + lh];
            const float b1 = Kl[(s0 + 1) * 68 + jt * 16 + lh];
            acc = wmma4(a0, a1, b0, b1, acc);
        }
        #pragma unroll
        for (int i = 0; i < 8; ++i) {
            const int row  = i + rofs;
            const int colk = jt * 16 + lh;
            const size_t gi = (rowg0 + row) * D_MODEL + h * DHEAD + colk;
            ob[gi] += acc[i] * Dtl[row * 68 + colk];
        }
    }
}

// ---------------------------------------------------------------------------
// GroupNorm (per 64-wide head) + SiLU gate
// ---------------------------------------------------------------------------
__global__ __launch_bounds__(32)
void rwkv_gnorm_gate(const float* __restrict__ o, const float* __restrict__ g,
                     const float* __restrict__ gnw, const float* __restrict__ gnb,
                     float* __restrict__ gated)
{
    const int mh   = blockIdx.x;
    const int row  = mh >> 4;
    const int h    = mh & 15;
    const int lane = threadIdx.x;
    const size_t rb = (size_t)row * D_MODEL + h * DHEAD;
    const float o0 = o[rb + lane];
    const float o1 = o[rb + lane + 32];
    float sum = o0 + o1, sq = o0 * o0 + o1 * o1;
    #pragma unroll
    for (int m = 1; m < 32; m <<= 1) {
        sum += __shfl_xor(sum, m, 32);
        sq  += __shfl_xor(sq,  m, 32);
    }
    const float mu  = sum * (1.0f / 64.0f);
    const float var = sq * (1.0f / 64.0f) - mu * mu;
    const float inv = rsqrtf(var + 0.00064f);
    const int d0 = h * DHEAD + lane, d1 = d0 + 32;
    gated[rb + lane]      = ((o0 - mu) * inv * gnw[d0] + gnb[d0]) * g[rb + lane];
    gated[rb + lane + 32] = ((o1 - mu) * inv * gnw[d1] + gnb[d1]) * g[rb + lane + 32];
}

extern "C" void kernel_launch(void* const* d_in, const int* in_sizes, int n_in,
                              void* d_out, int out_size, void* d_ws, size_t ws_size,
                              hipStream_t stream)
{
    (void)in_sizes; (void)n_in; (void)out_size; (void)ws_size;
    const float* x        = (const float*)d_in[0];
    const float* ddlerp_w = (const float*)d_in[1];
    const float* mix_r    = (const float*)d_in[2];
    const float* mix_k    = (const float*)d_in[3];
    const float* mix_v    = (const float*)d_in[4];
    const float* mix_g    = (const float*)d_in[5];
    const float* mix_w    = (const float*)d_in[6];
    const float* r_w      = (const float*)d_in[7];
    const float* k_w      = (const float*)d_in[8];
    const float* v_w      = (const float*)d_in[9];
    const float* g_w      = (const float*)d_in[10];
    const float* w_w      = (const float*)d_in[11];
    const float* out_w    = (const float*)d_in[12];
    const float* gnw      = (const float*)d_in[13];
    const float* gnb      = (const float*)d_in[14];

    const size_t SZ = (size_t)NROWS * D_MODEL;
    const size_t WSZ = (size_t)D_MODEL * D_MODEL;   // 1M elements per weight
    float* wsf = (float*)d_ws;
    float* dd   = wsf;
    float* rb_  = wsf + 1 * SZ;
    float* kb_  = wsf + 2 * SZ;
    float* vb_  = wsf + 3 * SZ;
    float* wb_  = wsf + 4 * SZ;
    float* gb_  = wsf + 5 * SZ;
    float* ob_  = wsf + 6 * SZ;
    float* dt_  = wsf + 7 * SZ;   // NTASK*16*64 == SZ (also hosts f16 weights pre-prep)
    float* kp_  = wsf + 8 * SZ;
    float* de_  = wsf + 9 * SZ;   // NTASK*64
    float* gated = dd;

    // f16 weight copies: 6 projection weights overlap dt_ (consumed before
    // rwkv_wkv_prep overwrites it); out_w copy lives after de_ (used last).
    _Float16* wh6   = (_Float16*)dt_;
    _Float16* whdd  = wh6 + 0 * WSZ;
    _Float16* whr   = wh6 + 1 * WSZ;
    _Float16* whk   = wh6 + 2 * WSZ;
    _Float16* whv   = wh6 + 3 * WSZ;
    _Float16* whg   = wh6 + 4 * WSZ;
    _Float16* whw   = wh6 + 5 * WSZ;
    _Float16* whout = (_Float16*)(de_ + (size_t)NTASK * DHEAD);

    const int cblk = (int)(WSZ / 4 / 256);   // 1024 blocks
    conv_f32_to_f16<<<cblk, 256, 0, stream>>>(ddlerp_w, whdd);
    conv_f32_to_f16<<<cblk, 256, 0, stream>>>(r_w, whr);
    conv_f32_to_f16<<<cblk, 256, 0, stream>>>(k_w, whk);
    conv_f32_to_f16<<<cblk, 256, 0, stream>>>(v_w, whv);
    conv_f32_to_f16<<<cblk, 256, 0, stream>>>(g_w, whg);
    conv_f32_to_f16<<<cblk, 256, 0, stream>>>(w_w, whw);
    conv_f32_to_f16<<<cblk, 256, 0, stream>>>(out_w, whout);

    dim3 gg(D_MODEL / 64, NROWS / 64), gt(128);
    rwkv_gemm_wmma<<<gg, gt, 0, stream>>>(x, nullptr, nullptr, whdd, dd, 0, 1);
    rwkv_gemm_wmma<<<gg, gt, 0, stream>>>(x, dd, mix_r, whr, rb_, 1, 0);
    rwkv_gemm_wmma<<<gg, gt, 0, stream>>>(x, dd, mix_k, whk, kb_, 1, 0);
    rwkv_gemm_wmma<<<gg, gt, 0, stream>>>(x, dd, mix_v, whv, vb_, 1, 0);
    rwkv_gemm_wmma<<<gg, gt, 0, stream>>>(x, dd, mix_g, whg, gb_, 1, 2);
    rwkv_gemm_wmma<<<gg, gt, 0, stream>>>(x, dd, mix_w, whw, wb_, 1, 3);
    rwkv_wkv_prep<<<NTASK, 64, 0, stream>>>(wb_, kb_, kp_, dt_, de_);
    rwkv_wkv_state<<<32, 256, 0, stream>>>(rb_, vb_, kp_, dt_, de_, ob_);
    rwkv_wkv_intra<<<NTASK, 32, 0, stream>>>(rb_, vb_, kp_, dt_, ob_);
    rwkv_gnorm_gate<<<NROWS * H_HEADS, 32, 0, stream>>>(ob_, gb_, gnw, gnb, gated);
    rwkv_gemm_wmma<<<gg, gt, 0, stream>>>(gated, nullptr, nullptr, whout, (float*)d_out, 2, 0);
}
